// PGCBlock_18751827214385
// MI455X (gfx1250) — matrix operand order repaired and checked by reading, not compiled, not run on hardware
//
#include <hip/hip_runtime.h>
#include <hip/hip_bf16.h>

// ---------------------------------------------------------------------------
// PGC block for MI455X (gfx1250, wave32, WMMA bf16 16x16x32).
//   K0: convert w_in / w_out to bf16 in workspace (no cvt in GEMM hot loops,
//       half the A-operand bytes)
//   K1: h = w_in @ x + b_in  -> bf16 workspace, + per-column sum(h^2)
//   K2: s1 = rsqrt(mean+eps); g = (dwconv(xc*s1*g1)+b_dw)*(v*s1*g1);
//       o = w_out @ g + b_out; out = x + rmsnorm(o)*g2
// ---------------------------------------------------------------------------

#define BB   8
#define DD   512
#define LL   4096
#define HH   1024
#define H2   2048
#define KKC  7
#define DILC 2
#define PADC 6
#define EPSV 1.1920929e-07f

typedef __bf16 bf16_t;
typedef __bf16 v16bf __attribute__((ext_vector_type(16)));
typedef __bf16 v8bf  __attribute__((ext_vector_type(8)));
typedef __bf16 v2bf  __attribute__((ext_vector_type(2)));
typedef float  v8f   __attribute__((ext_vector_type(8)));

static __device__ inline v16bf cat16(v8bf lo, v8bf hi) {
  return __builtin_shufflevector(lo, hi, 0,1,2,3,4,5,6,7,8,9,10,11,12,13,14,15);
}

// A fragment (16x32 bf16) from a row-major bf16 matrix in global memory.
// ISA layout: lanes 0-15 row m=lane hold K runs [0,8) and [16,24);
//             lanes 16-31 row m=lane-16 hold K runs [8,16) and [24,32).
static __device__ inline v16bf load_a_frag_bf16(const bf16_t* __restrict__ base,
                                                int ld) {
  const int lane = (int)(threadIdx.x & 31u);
  const int m  = lane & 15;
  const int k0 = (lane >> 4) << 3;          // 0 or 8
  const bf16_t* p = base + (size_t)m * ld + k0;
  v8bf lo = *(const v8bf*)(p);              // K run [k0, k0+8)
  v8bf hi = *(const v8bf*)(p + 16);         // K run [k0+16, k0+24)
  return cat16(lo, hi);
}

// B fragment (32x16 bf16) from LDS stored column-major: row n (N position),
// contiguous K along the row, row stride ldk (bf16 elems, padded).
// ISA layout: lanes 0-15 col n=lane hold K runs [0,8) and [16,24);
//             lanes 16-31 hold K runs [8,16) and [24,32).
static __device__ inline v16bf load_b_frag_lds(const bf16_t* base, int ldk) {
  const int lane = (int)(threadIdx.x & 31u);
  const int n  = lane & 15;
  const int k0 = (lane >> 4) << 3;          // 0 or 8
  const bf16_t* p = base + n * ldk + k0;
  v8bf lo = *(const v8bf*)(p);
  v8bf hi = *(const v8bf*)(p + 16);
  return cat16(lo, hi);
}

// ---------------------------------------------------------------------------
// K0: one-shot weight conversion f32 -> bf16 (w_in: 2048x512, w_out: 512x1024)
// ---------------------------------------------------------------------------
__global__ __launch_bounds__(256)
void k0_cvt_weights(const float* __restrict__ w_in,
                    const float* __restrict__ w_out,
                    bf16_t* __restrict__ w_in_bf,
                    bf16_t* __restrict__ w_out_bf) {
  const int i = (int)(blockIdx.x * 256 + threadIdx.x);
  if (i < H2 * DD) w_in_bf[i] = (bf16_t)w_in[i];
  if (i < DD * HH) w_out_bf[i] = (bf16_t)w_out[i];
}

// ---------------------------------------------------------------------------
// K1: GEMM1 (2048x512 @ 512x4096 per batch), bf16 output + column sum-of-sq.
// Grid: (L/128, 2048/256, B). Block: 512 threads (16 waves, 4x4 wave grid).
// ---------------------------------------------------------------------------
__global__ __launch_bounds__(512)
void k1_gemm_in(const float* __restrict__ x, const bf16_t* __restrict__ w_in,
                const float* __restrict__ b_in, bf16_t* __restrict__ h_ws,
                float* __restrict__ sumsq1) {
  __shared__ bf16_t Bs[128 * 72];           // x tile, [l][d], K-chunk 64, +8 pad
  __shared__ float  colsq[128];

  const int tid  = (int)threadIdx.x;
  const int lane = tid & 31;
  const int w    = tid >> 5;
  const int wm   = w >> 2;                  // 0..3
  const int wn   = w & 3;                   // 0..3
  const int b    = (int)blockIdx.z;
  const int mblk = (int)blockIdx.y * 256;
  const int l0   = (int)blockIdx.x * 128;

  if (tid < 128) colsq[tid] = 0.0f;

  v8f acc[4][2];
#pragma unroll
  for (int i = 0; i < 4; ++i)
#pragma unroll
    for (int j = 0; j < 2; ++j)
#pragma unroll
      for (int r = 0; r < 8; ++r) acc[i][j][r] = 0.0f;

  const float* xb = x + (size_t)b * DD * LL;

  for (int k0 = 0; k0 < DD; k0 += 64) {
    __syncthreads();                        // protect Bs (and colsq init)
#pragma unroll
    for (int i = 0; i < 8; ++i) {
      int pidx = i * 512 + tid;             // 32 x 128 (d-pair, l) elements
      int d2 = pidx >> 7;                   // 0..31 -> d = 2*d2
      int l  = pidx & 127;
      float a0 = xb[(size_t)(k0 + 2 * d2) * LL + (l0 + l)];
      float a1 = xb[(size_t)(k0 + 2 * d2 + 1) * LL + (l0 + l)];
      v2bf pk; pk[0] = (bf16_t)a0; pk[1] = (bf16_t)a1;
      *(v2bf*)(&Bs[l * 72 + 2 * d2]) = pk;  // one ds_store_b32
    }
    __syncthreads();

#pragma unroll
    for (int kk = 0; kk < 64; kk += 32) {
      v16bf bf[2];
#pragma unroll
      for (int j = 0; j < 2; ++j)
        bf[j] = load_b_frag_lds(Bs + (wn * 32 + j * 16) * 72 + kk, 72);
#pragma unroll
      for (int i = 0; i < 4; ++i) {
        v16bf af = load_a_frag_bf16(
            w_in + (size_t)(mblk + wm * 64 + i * 16) * DD + (k0 + kk), DD);
#pragma unroll
        for (int j = 0; j < 2; ++j)
          acc[i][j] = __builtin_amdgcn_wmma_f32_16x16x32_bf16(
              false, af, false, bf[j], (short)0, acc[i][j], false, false);
      }
    }
  }
  __syncthreads();

  // Epilogue: +bias, store bf16 pre-norm h, reduce column sum-of-squares.
  const int mro = 8 * (lane >> 4);
#pragma unroll
  for (int j = 0; j < 2; ++j) {
    float ss = 0.0f;
    const int l = l0 + wn * 32 + j * 16 + (lane & 15);
#pragma unroll
    for (int i = 0; i < 4; ++i) {
      const int mb = mblk + wm * 64 + i * 16;
#pragma unroll
      for (int r = 0; r < 8; ++r) {
        const int m = mb + r + mro;
        float v = acc[i][j][r] + b_in[m];
        ss += v * v;
        h_ws[((size_t)b * H2 + m) * LL + l] = (bf16_t)v;
      }
    }
    atomicAdd(&colsq[wn * 32 + j * 16 + (lane & 15)], ss);
  }
  __syncthreads();
  if (tid < 128) atomicAdd(&sumsq1[(size_t)b * LL + l0 + tid], colsq[tid]);
}

// ---------------------------------------------------------------------------
// K2: fused norm1-apply + dilated depthwise conv + gate + GEMM2 + norm2 +
// residual. Grid: (L/64, B). Block: 512 threads (16 waves, 8x2 wave grid).
// ---------------------------------------------------------------------------
__global__ __launch_bounds__(512)
void k2_fused(const float* __restrict__ x, const bf16_t* __restrict__ h_ws,
              const float* __restrict__ sumsq1, const float* __restrict__ g1,
              const float* __restrict__ w_dw, const float* __restrict__ b_dw,
              const bf16_t* __restrict__ w_out, const float* __restrict__ b_out,
              const float* __restrict__ g2, float* __restrict__ out) {
  __shared__ bf16_t Gs[64 * 136];           // g chunk, [n][c], +8 pad
  __shared__ float  s1s[76];                // rsqrt scales for halo window
  __shared__ float  colsq[64];

  const int tid  = (int)threadIdx.x;
  const int lane = tid & 31;
  const int w    = tid >> 5;
  const int wm   = w >> 1;                  // 0..7
  const int wn   = w & 1;                   // 0..1
  const int b    = (int)blockIdx.y;
  const int l0   = (int)blockIdx.x * 64;

  if (tid < 76) {
    int l = l0 - PADC + tid;
    float s = 0.0f;
    if (l >= 0 && l < LL)
      s = rsqrtf(sumsq1[(size_t)b * LL + l] * (1.0f / (float)H2) + EPSV);
    s1s[tid] = s;
  }
  if (tid >= 448) colsq[tid - 448] = 0.0f;

  v8f acc[4][2];
#pragma unroll
  for (int i = 0; i < 4; ++i)
#pragma unroll
    for (int j = 0; j < 2; ++j)
#pragma unroll
      for (int r = 0; r < 8; ++r) acc[i][j][r] = 0.0f;

  const int n  = tid & 63;
  const int cb = tid >> 6;                  // 0..7
  __syncthreads();
  const float s1c = s1s[n + PADC];

  for (int ch0 = 0; ch0 < HH; ch0 += 128) {
    // Build the 128-channel gate chunk into LDS (bf16, WMMA-B layout),
    // two adjacent channels per thread -> packed b32 LDS stores.
    for (int t = 0; t < 8; ++t) {
      const int c0 = ch0 + 2 * cb + 16 * t;
      v2bf pk;
#pragma unroll
      for (int u = 0; u < 2; ++u) {
        const int c = c0 + u;
        float a = b_dw[c];
        const float g1c = g1[c];
        const bf16_t* hrow = h_ws + ((size_t)b * H2 + c) * LL;
#pragma unroll
        for (int j = 0; j < KKC; ++j) {
          const int l = l0 + n + DILC * j - PADC;
          float xv = 0.0f;
          if ((unsigned)l < (unsigned)LL)
            xv = (float)hrow[l] * s1s[n + DILC * j] * g1c;
          a += w_dw[c * KKC + j] * xv;
        }
        const float vv =
            (float)h_ws[((size_t)b * H2 + HH + c) * LL + (l0 + n)] * s1c *
            g1[HH + c];
        pk[u] = (bf16_t)(a * vv);
      }
      *(v2bf*)(&Gs[n * 136 + (c0 - ch0)]) = pk;
    }
    __syncthreads();

    // GEMM2 partial over this 128-wide K chunk.
#pragma unroll
    for (int kl = 0; kl < 128; kl += 32) {
      v16bf bf[2];
#pragma unroll
      for (int j = 0; j < 2; ++j)
        bf[j] = load_b_frag_lds(Gs + (wn * 32 + j * 16) * 136 + kl, 136);
#pragma unroll
      for (int i = 0; i < 4; ++i) {
        v16bf af = load_a_frag_bf16(
            w_out + (size_t)(wm * 64 + i * 16) * HH + (ch0 + kl), HH);
#pragma unroll
        for (int j = 0; j < 2; ++j)
          acc[i][j] = __builtin_amdgcn_wmma_f32_16x16x32_bf16(
              false, af, false, bf[j], (short)0, acc[i][j], false, false);
      }
    }
    __syncthreads();
  }

  // Epilogue: +bias, norm2 reduction (full M=512 in this block), residual.
  const int mro = 8 * (lane >> 4);
#pragma unroll
  for (int j = 0; j < 2; ++j) {
    float ss = 0.0f;
#pragma unroll
    for (int i = 0; i < 4; ++i) {
#pragma unroll
      for (int r = 0; r < 8; ++r) {
        const int m = wm * 64 + i * 16 + r + mro;
        float v = acc[i][j][r] + b_out[m];
        acc[i][j][r] = v;
        ss += v * v;
      }
    }
    atomicAdd(&colsq[wn * 32 + j * 16 + (lane & 15)], ss);
  }
  __syncthreads();
#pragma unroll
  for (int j = 0; j < 2; ++j) {
    const int nn = wn * 32 + j * 16 + (lane & 15);
    const int l = l0 + nn;
    const float s2 = rsqrtf(colsq[nn] * (1.0f / (float)DD) + EPSV);
#pragma unroll
    for (int i = 0; i < 4; ++i) {
#pragma unroll
      for (int r = 0; r < 8; ++r) {
        const int m = wm * 64 + i * 16 + r + mro;
        const size_t oi = ((size_t)b * DD + m) * LL + l;
        out[oi] = x[oi] + acc[i][j][r] * s2 * g2[m];
      }
    }
  }
}

// ---------------------------------------------------------------------------
extern "C" void kernel_launch(void* const* d_in, const int* in_sizes, int n_in,
                              void* d_out, int out_size, void* d_ws, size_t ws_size,
                              hipStream_t stream) {
  (void)in_sizes; (void)n_in; (void)out_size; (void)ws_size;
  const float* x     = (const float*)d_in[0];
  const float* w_in  = (const float*)d_in[1];
  const float* b_in  = (const float*)d_in[2];
  const float* g1    = (const float*)d_in[3];
  const float* w_dw  = (const float*)d_in[4];
  const float* b_dw  = (const float*)d_in[5];
  const float* w_out = (const float*)d_in[6];
  const float* b_out = (const float*)d_in[7];
  const float* g2    = (const float*)d_in[8];
  float* out = (float*)d_out;

  // Workspace layout:
  //   [sumsq1: B*L f32][w_in_bf: 2H*D bf16][w_out_bf: D*H bf16][h_ws: B*2H*L bf16]
  char* ws = (char*)d_ws;
  float*  sumsq1   = (float*)ws;                 ws += (size_t)BB * LL * sizeof(float);
  bf16_t* w_in_bf  = (bf16_t*)ws;                ws += (size_t)H2 * DD * sizeof(bf16_t);
  bf16_t* w_out_bf = (bf16_t*)ws;                ws += (size_t)DD * HH * sizeof(bf16_t);
  bf16_t* h_ws     = (bf16_t*)ws;

  hipMemsetAsync(sumsq1, 0, (size_t)BB * LL * sizeof(float), stream);

  k0_cvt_weights<<<dim3((H2 * DD) / 256), dim3(256), 0, stream>>>(
      w_in, w_out, w_in_bf, w_out_bf);

  dim3 grid1(LL / 128, H2 / 256, BB);       // (32, 8, 8)
  k1_gemm_in<<<grid1, dim3(512), 0, stream>>>(x, w_in_bf, b_in, h_ws, sumsq1);

  dim3 grid2(LL / 64, BB);                  // (64, 8)
  k2_fused<<<grid2, dim3(512), 0, stream>>>(x, h_ws, sumsq1, g1, w_dw, b_dw,
                                            w_out_bf, b_out, g2, out);
}
